// QuantumGate_53214644797979
// MI455X (gfx1250) — matrix operand order repaired and checked by reading, not compiled, not run on hardware
//
#include <hip/hip_runtime.h>

typedef __attribute__((ext_vector_type(16))) _Float16 v16h;
typedef __attribute__((ext_vector_type(8)))  _Float16 v8h;
typedef __attribute__((ext_vector_type(8)))  float    v8f;

#define NB   8192
#define KTOT 1024
#define NQ   8
#define HIDN 512
#define PI_F 3.14159265358979323846f

#define A_THREADS 128           // 4 waves -> 64 rows per block
#define A_ROWS    (4 * 16)

// ---------------------------------------------------------------------------
// Kernel A: q_in[B,8] = pi * tanh(concat(x,h) @ W_in^T + b_in) via WMMA f16.
// W_in staged in LDS as f16 (16 rows x 1024, rows 8..15 zero) once per block.
// One wave per 16-row tile; 4 waves (64 rows) per block, 128 blocks.
// ---------------------------------------------------------------------------
__device__ __forceinline__ v8f gemm_k512(v8f acc, const float* __restrict__ arow,
                                         const _Float16* __restrict__ wrow,
                                         int half) {
  for (int k0 = 0; k0 < 512; k0 += 32) {
    // A: 16x32 f16 fragment. lane<16: K = k0+[0..7], k0+[16..23];
    //                        lane>=16: K = k0+[8..15], k0+[24..31]
    const float* sa = arow + k0 + half * 8;
    float4 f0 = *(const float4*)(sa);
    float4 f1 = *(const float4*)(sa + 4);
    float4 f2 = *(const float4*)(sa + 16);
    float4 f3 = *(const float4*)(sa + 20);
    v16h a;
    a[0]  = (_Float16)f0.x; a[1]  = (_Float16)f0.y;
    a[2]  = (_Float16)f0.z; a[3]  = (_Float16)f0.w;
    a[4]  = (_Float16)f1.x; a[5]  = (_Float16)f1.y;
    a[6]  = (_Float16)f1.z; a[7]  = (_Float16)f1.w;
    a[8]  = (_Float16)f2.x; a[9]  = (_Float16)f2.y;
    a[10] = (_Float16)f2.z; a[11] = (_Float16)f2.w;
    a[12] = (_Float16)f3.x; a[13] = (_Float16)f3.y;
    a[14] = (_Float16)f3.z; a[15] = (_Float16)f3.w;

    // B: 32x16 f16 fragment from LDS (unconditional ds_load_b128 x2)
    const _Float16* sb = wrow + k0 + half * 8;
    v8h b0 = *(const v8h*)(sb);
    v8h b1 = *(const v8h*)(sb + 16);
    v16h b;
#pragma unroll
    for (int j = 0; j < 8; ++j) { b[j] = b0[j]; b[8 + j] = b1[j]; }

    acc = __builtin_amdgcn_wmma_f32_16x16x32_f16(false, a, false, b,
                                                 (short)0, acc, false, false);
  }
  return acc;
}

__global__ __launch_bounds__(A_THREADS) void qin_wmma_kernel(
    const float* __restrict__ x_t, const float* __restrict__ h_prev,
    const float* __restrict__ W_in, const float* __restrict__ b_in,
    float* __restrict__ q_in) {
  __shared__ _Float16 wlds[16 * KTOT];          // 32 KB; rows 8..15 are zero

  const int tid = threadIdx.x;
  // Fill rows 0..7 from W_in (f32 -> f16), coalesced float4 reads.
  for (int i = tid * 4; i < NQ * KTOT; i += A_THREADS * 4) {
    float4 f = *(const float4*)(W_in + i);
    wlds[i + 0] = (_Float16)f.x;
    wlds[i + 1] = (_Float16)f.y;
    wlds[i + 2] = (_Float16)f.z;
    wlds[i + 3] = (_Float16)f.w;
  }
  // Zero rows 8..15 (exact 4-half stores, in-bounds).
  for (int i = tid * 4; i < NQ * KTOT; i += A_THREADS * 4) {
    wlds[NQ * KTOT + i + 0] = (_Float16)0.0f;
    wlds[NQ * KTOT + i + 1] = (_Float16)0.0f;
    wlds[NQ * KTOT + i + 2] = (_Float16)0.0f;
    wlds[NQ * KTOT + i + 3] = (_Float16)0.0f;
  }
  __syncthreads();

  const int lane = tid & 31;
  const int wave = tid >> 5;
  const int rt   = (blockIdx.x * (A_THREADS / 32) + wave) * 16;
  const int half = lane >> 4;
  const int m16  = lane & 15;

  const int rowA = rt + m16;
  const float* xrow = x_t    + (size_t)rowA * 512;
  const float* hrow = h_prev + (size_t)rowA * 512;
  const _Float16* wrow = wlds + (size_t)m16 * KTOT;

  v8f acc = {};
  acc = gemm_k512(acc, xrow, wrow,       half);  // K = 0..511
  acc = gemm_k512(acc, hrow, wrow + 512, half);  // K = 512..1023

  if (m16 < NQ) {
    const float bn = b_in[m16];
#pragma unroll
    for (int r = 0; r < 8; ++r) {
      const int row = rt + half * 8 + r;        // C/D layout: M = r + half*8
      q_in[row * NQ + m16] = PI_F * tanhf(acc[r] + bn);
    }
  }
}

// ---------------------------------------------------------------------------
// Kernel B: 8-qubit state-vector VQC, one wave32 per sample.
// amp index = lane*8 + k ; wire q acts on bit P = 7-q.
// P>=3 -> lane bit (P-3): cross-lane shuffles. P<3 -> in-register k-bit pairs.
// ---------------------------------------------------------------------------
__device__ __forceinline__ float wsx(float v, int mask) {
  return __shfl_xor(v, mask, 32);
}

template <int P>
__device__ __forceinline__ void apply_rx(float (&sr)[8], float (&si)[8],
                                         float c, float s, int lane) {
  if constexpr (P >= 3) {
    const int mask = 1 << (P - 3);
#pragma unroll
    for (int k = 0; k < 8; ++k) {
      float pr = wsx(sr[k], mask), pi = wsx(si[k], mask);
      float nr = c * sr[k] + s * pi;              // (-i s)*partner
      float ni = c * si[k] - s * pr;
      sr[k] = nr; si[k] = ni;
    }
  } else {
#pragma unroll
    for (int k0 = 0; k0 < 8; ++k0) {
      if ((k0 >> P) & 1) continue;
      const int k1 = k0 + (1 << P);
      float a0r = sr[k0], a0i = si[k0], a1r = sr[k1], a1i = si[k1];
      sr[k0] = c * a0r + s * a1i;  si[k0] = c * a0i - s * a1r;
      sr[k1] = c * a1r + s * a0i;  si[k1] = c * a1i - s * a0r;
    }
  }
}

template <int P>
__device__ __forceinline__ void apply_ry(float (&sr)[8], float (&si)[8],
                                         float c, float s, int lane) {
  if constexpr (P >= 3) {
    const int mask = 1 << (P - 3);
    const int bit  = (lane >> (P - 3)) & 1;
    const float sg = bit ? s : -s;
#pragma unroll
    for (int k = 0; k < 8; ++k) {
      float pr = wsx(sr[k], mask), pi = wsx(si[k], mask);
      sr[k] = c * sr[k] + sg * pr;
      si[k] = c * si[k] + sg * pi;
    }
  } else {
#pragma unroll
    for (int k0 = 0; k0 < 8; ++k0) {
      if ((k0 >> P) & 1) continue;
      const int k1 = k0 + (1 << P);
      float a0r = sr[k0], a0i = si[k0], a1r = sr[k1], a1i = si[k1];
      sr[k0] = c * a0r - s * a1r;  si[k0] = c * a0i - s * a1i;
      sr[k1] = s * a0r + c * a1r;  si[k1] = s * a0i + c * a1i;
    }
  }
}

template <int P>
__device__ __forceinline__ void apply_rz(float (&sr)[8], float (&si)[8],
                                         float cb, float sb, int lane) {
#pragma unroll
  for (int k = 0; k < 8; ++k) {
    int bit;
    if constexpr (P >= 3) bit = (lane >> (P - 3)) & 1;
    else                  bit = (k >> P) & 1;
    const float sg = bit ? sb : -sb;
    float nr = cb * sr[k] - sg * si[k];
    float ni = cb * si[k] + sg * sr[k];
    sr[k] = nr; si[k] = ni;
  }
}

template <int PC, int PT>
__device__ __forceinline__ void cnot_g(float (&sr)[8], float (&si)[8], int lane) {
  if constexpr (PT >= 3) {
    const int mask = 1 << (PT - 3);
#pragma unroll
    for (int k = 0; k < 8; ++k) {
      float pr = wsx(sr[k], mask), pi = wsx(si[k], mask);
      int cbit;
      if constexpr (PC >= 3) cbit = (lane >> (PC - 3)) & 1;
      else                   cbit = (k >> PC) & 1;
      sr[k] = cbit ? pr : sr[k];
      si[k] = cbit ? pi : si[k];
    }
  } else if constexpr (PC >= 3) {
    const int cbit = (lane >> (PC - 3)) & 1;
#pragma unroll
    for (int k0 = 0; k0 < 8; ++k0) {
      if ((k0 >> PT) & 1) continue;
      const int k1 = k0 + (1 << PT);
      float t0r = sr[k0], t0i = si[k0];
      sr[k0] = cbit ? sr[k1] : sr[k0];  si[k0] = cbit ? si[k1] : si[k0];
      sr[k1] = cbit ? t0r    : sr[k1];  si[k1] = cbit ? t0i    : si[k1];
    }
  } else {
#pragma unroll
    for (int k0 = 0; k0 < 8; ++k0) {
      if ((k0 >> PT) & 1) continue;
      if (!((k0 >> PC) & 1)) continue;
      const int k1 = k0 + (1 << PT);
      float t = sr[k0]; sr[k0] = sr[k1]; sr[k1] = t;
      t = si[k0]; si[k0] = si[k1]; si[k1] = t;
    }
  }
}

template <int P>
__device__ __forceinline__ float ev_wire(const float (&p)[8], float tot, int lane) {
  float ev;
  if constexpr (P >= 3) {
    ev = ((lane >> (P - 3)) & 1) ? -tot : tot;
  } else {
    ev = 0.0f;
#pragma unroll
    for (int k = 0; k < 8; ++k) ev += ((k >> P) & 1) ? -p[k] : p[k];
  }
#pragma unroll
  for (int m = 16; m >= 1; m >>= 1) ev += wsx(ev, m);
  return ev;
}

__global__ __launch_bounds__(256) void vqc_kernel(
    const float* __restrict__ q_in, const float* __restrict__ vw,
    float* __restrict__ q_feat) {
  const int lane   = threadIdx.x & 31;
  const int sample = blockIdx.x * 8 + (threadIdx.x >> 5);

  float ang[8];
#pragma unroll
  for (int q = 0; q < 8; ++q) ang[q] = 0.5f * q_in[sample * NQ + q];

  float sr[8], si[8];
#pragma unroll
  for (int k = 0; k < 8; ++k) { sr[k] = 0.0625f; si[k] = 0.0f; }  // H^n |0>

#define DO_RX(q) { float c = __cosf(ang[q]), s = __sinf(ang[q]); \
                   apply_rx<7 - (q)>(sr, si, c, s, lane); }
  DO_RX(0) DO_RX(1) DO_RX(2) DO_RX(3) DO_RX(4) DO_RX(5) DO_RX(6) DO_RX(7)

#define DO_RYRZ(q) { float a = 0.5f * vw[q];       \
                     apply_ry<7 - (q)>(sr, si, __cosf(a), __sinf(a), lane); \
                     float b = 0.5f * vw[8 + (q)]; \
                     apply_rz<7 - (q)>(sr, si, __cosf(b), __sinf(b), lane); }
  DO_RYRZ(0) DO_RYRZ(1) DO_RYRZ(2) DO_RYRZ(3)
  DO_RYRZ(4) DO_RYRZ(5) DO_RYRZ(6) DO_RYRZ(7)

  cnot_g<7, 6>(sr, si, lane);
  cnot_g<6, 5>(sr, si, lane);
  cnot_g<5, 4>(sr, si, lane);
  cnot_g<4, 3>(sr, si, lane);
  cnot_g<3, 2>(sr, si, lane);
  cnot_g<2, 1>(sr, si, lane);
  cnot_g<1, 0>(sr, si, lane);
  cnot_g<0, 7>(sr, si, lane);

  float p[8];
  float tot = 0.0f;
#pragma unroll
  for (int k = 0; k < 8; ++k) { p[k] = sr[k] * sr[k] + si[k] * si[k]; tot += p[k]; }

#define DO_EV(q) { float ev = ev_wire<7 - (q)>(p, tot, lane); \
                   if (lane == 0) q_feat[sample * NQ + (q)] = ev; }
  DO_EV(0) DO_EV(1) DO_EV(2) DO_EV(3) DO_EV(4) DO_EV(5) DO_EV(6) DO_EV(7)
}

// ---------------------------------------------------------------------------
// Kernel C: out[B,512] = q_feat[B,8] @ W_out^T + b_out   (memory-bound, K=8)
// ---------------------------------------------------------------------------
__global__ __launch_bounds__(256) void out_kernel(
    const float* __restrict__ q_feat, const float* __restrict__ W_out,
    const float* __restrict__ b_out, float* __restrict__ out) {
  const int idx = blockIdx.x * blockDim.x + threadIdx.x;
  const int r = idx >> 9;
  const int c = idx & (HIDN - 1);
  float v = b_out[c];
#pragma unroll
  for (int k = 0; k < 8; ++k) v += q_feat[r * NQ + k] * W_out[c * NQ + k];
  out[idx] = v;
}

// ---------------------------------------------------------------------------
extern "C" void kernel_launch(void* const* d_in, const int* in_sizes, int n_in,
                              void* d_out, int out_size, void* d_ws, size_t ws_size,
                              hipStream_t stream) {
  const float* x_t    = (const float*)d_in[0];
  const float* h_prev = (const float*)d_in[1];
  const float* W_in   = (const float*)d_in[2];
  const float* b_in   = (const float*)d_in[3];
  const float* vw     = (const float*)d_in[4];
  const float* W_out  = (const float*)d_in[5];
  const float* b_out  = (const float*)d_in[6];
  float* out = (float*)d_out;

  float* q_in   = (float*)d_ws;                 // [8192, 8]
  float* q_feat = q_in + (size_t)NB * NQ;       // [8192, 8]

  qin_wmma_kernel<<<NB / A_ROWS, A_THREADS, 0, stream>>>(x_t, h_prev, W_in, b_in, q_in);
  vqc_kernel<<<NB / 8, 256, 0, stream>>>(q_in, vw, q_feat);
  out_kernel<<<(NB * HIDN) / 256, 256, 0, stream>>>(q_feat, W_out, b_out, out);
}
